// KNNLayer_74586402062895
// MI455X (gfx1250) — matrix operand order repaired and checked by reading, not compiled, not run on hardware
//
#include <hip/hip_runtime.h>
#include <hip/hip_bf16.h>

typedef __attribute__((ext_vector_type(16))) __bf16 v16bf;
typedef __attribute__((ext_vector_type(8)))  float  v8f;

#define NPTS       100      // real reference points
#define NPAD       112      // padded to 7 tiles of 16
#define NT_TILES   7        // 112 / 16
#define DIM        256
#define TOPK       5
#define ROWS       16384
#define BIGF       3.0e38f

// LDS layout (dynamic): bf16-hi plane, bf16-lo plane, rnorm
#define SHI_BYTES  (NPAD * DIM * 2)           // 57344
#define SLO_OFF    SHI_BYTES
#define RN_OFF     (2 * SHI_BYTES)            // 114688
#define SMEM_BYTES (2 * SHI_BYTES + NPAD * 4) // 115136

__global__ void
__attribute__((amdgpu_flat_work_group_size(256, 256), amdgpu_waves_per_eu(1)))
knn_topk_kernel(const float* __restrict__ x,
                const float* __restrict__ ref,
                int* __restrict__ out)
{
    extern __shared__ __align__(32) unsigned char smem[];
    unsigned short* sHi = (unsigned short*)smem;
    unsigned short* sLo = (unsigned short*)(smem + SLO_OFF);
    float*          sRn = (float*)(smem + RN_OFF);

    const int tid    = threadIdx.x;
    const int lane   = tid & 31;
    const int wave   = tid >> 5;
    const int l15    = lane & 15;
    const int hiHalf = (lane >> 4) & 1;          // 0 for lanes 0-15, 1 for 16-31

    // ---------------- Stage ref -> LDS as bf16 hi/lo planes (once per block) ----
    for (int idx = tid; idx < NPAD * DIM; idx += 256) {
        const int r = idx >> 8;                  // DIM == 256
        const int c = idx & 255;
        float v = (r < NPTS) ? ref[r * DIM + c] : 0.0f;
        __bf16 h = (__bf16)v;
        __bf16 l = (__bf16)(v - (float)h);
        sHi[idx] = __builtin_bit_cast(unsigned short, h);
        sLo[idx] = __builtin_bit_cast(unsigned short, l);
    }
    // ||r||^2 (fp32 exact); padded rows pushed to +inf so top-k never picks them
    if (tid < NPAD) {
        float s = BIGF;
        if (tid < NPTS) {
            s = 0.0f;
            const float4* row = (const float4*)(ref + tid * DIM);
            #pragma unroll 4
            for (int c = 0; c < DIM / 4; ++c) {
                float4 f = row[c];
                s += f.x * f.x + f.y * f.y + f.z * f.z + f.w * f.w;
            }
        }
        sRn[tid] = s;
    }
    __syncthreads();

    // ---------------- GEMM: acc[nt] = x_tile (16x256) . ref_tile^T -------------
    const int mbase = blockIdx.x * 128 + wave * 16;     // 16 rows per wave
    const int mrow  = mbase + l15;                      // this lane's A row
    const float* xrow = x + (size_t)mrow * DIM;

    v8f acc[NT_TILES];
    #pragma unroll
    for (int nt = 0; nt < NT_TILES; ++nt) acc[nt] = (v8f){0,0,0,0,0,0,0,0};

    float xnp = 0.0f;                                   // partial ||x_row||^2

    for (int kt = 0; kt < DIM / 32; ++kt) {
        // A fragment: lane holds row (l15); lanes<16 take K [0,16),
        // lanes>=16 take [16,32) within this 32-wide K tile.
        const int kb = kt * 32 + hiHalf * 16;
        const float4 f0 = *(const float4*)(xrow + kb);
        const float4 f1 = *(const float4*)(xrow + kb + 4);
        const float4 f2 = *(const float4*)(xrow + kb + 8);
        const float4 f3 = *(const float4*)(xrow + kb + 12);

        v16bf a_hi, a_lo;
        // Convert in place (no float staging array -> minimal transient pressure)
        #define CVT(V, I) do { float _v = (V); xnp += _v * _v;            \
                               __bf16 _h = (__bf16)_v; a_hi[I] = _h;      \
                               a_lo[I] = (__bf16)(_v - (float)_h); } while (0)
        CVT(f0.x, 0);  CVT(f0.y, 1);  CVT(f0.z, 2);  CVT(f0.w, 3);
        CVT(f1.x, 4);  CVT(f1.y, 5);  CVT(f1.z, 6);  CVT(f1.w, 7);
        CVT(f2.x, 8);  CVT(f2.y, 9);  CVT(f2.z, 10); CVT(f2.w, 11);
        CVT(f3.x, 12); CVT(f3.y, 13); CVT(f3.z, 14); CVT(f3.w, 15);
        #undef CVT

        // B fragments from LDS: lane = column n = l15, K-half by lane>=16.
        const int kcol = kt * 32 + hiHalf * 16;         // 16 consecutive bf16 = 32B
        #pragma unroll
        for (int nt = 0; nt < NT_TILES; ++nt) {
            const int n   = nt * 16 + l15;
            const int off = n * DIM + kcol;             // ushort units, 32B aligned
            v16bf b_hi = *reinterpret_cast<const v16bf*>(sHi + off);
            v16bf b_lo = *reinterpret_cast<const v16bf*>(sLo + off);
            // bf16x3: hi*hi + hi*lo + lo*hi, f32 accumulate
            acc[nt] = __builtin_amdgcn_wmma_f32_16x16x32_bf16(
                false, a_hi, false, b_hi, (short)0, acc[nt], false, false);
            acc[nt] = __builtin_amdgcn_wmma_f32_16x16x32_bf16(
                false, a_hi, false, b_lo, (short)0, acc[nt], false, false);
            acc[nt] = __builtin_amdgcn_wmma_f32_16x16x32_bf16(
                false, a_lo, false, b_hi, (short)0, acc[nt], false, false);
        }
    }

    // ---------------- ||x||^2 redistribution ---------------------------------
    // lane L and L^16 jointly hold row (L&15): combine, then broadcast so each
    // lane gets the norms of the 8 C-rows it owns (rows rowBase..rowBase+7).
    float xnfull = xnp + __shfl_xor(xnp, 16, 32);       // row (l15) full norm
    const int rowBase = hiHalf * 8;                     // C rows: lanes<16 -> 0..7
    float xnrow[8];
    #pragma unroll
    for (int v = 0; v < 8; ++v)
        xnrow[v] = __shfl(xnfull, rowBase + v, 32);     // ds_bpermute

    float rn[NT_TILES];
    #pragma unroll
    for (int nt = 0; nt < NT_TILES; ++nt) rn[nt] = sRn[nt * 16 + l15];

    // ---------------- Top-5 per row (iterative argmin over 16 lanes x 7 regs) --
    #pragma unroll
    for (int v = 0; v < 8; ++v) {
        float dl[NT_TILES];
        #pragma unroll
        for (int nt = 0; nt < NT_TILES; ++nt)
            dl[nt] = xnrow[v] + rn[nt] - 2.0f * acc[nt][v];   // d^2 (sqrt monotonic)

        const int gRow = mbase + rowBase + v;
        for (int j = 0; j < TOPK; ++j) {
            float bv = dl[0];
            int   bi = l15;                              // n = l15 + 16*nt
            #pragma unroll
            for (int nt = 1; nt < NT_TILES; ++nt) {
                float c = dl[nt];
                int  ci = l15 + 16 * nt;
                if (c < bv) { bv = c; bi = ci; }         // nt ascending: lowest idx kept
            }
            // butterfly min over the 16-lane half; tie-break to lowest index
            #pragma unroll
            for (int m = 1; m < 16; m <<= 1) {
                float ov = __shfl_xor(bv, m, 32);
                int   oi = __shfl_xor(bi, m, 32);
                if (ov < bv || (ov == bv && oi < bi)) { bv = ov; bi = oi; }
            }
            if (l15 == j) out[gRow * TOPK + j] = bi;
            // invalidate winner slot (unrolled select -> no dynamic reg indexing)
            const int nsel = bi >> 4;
            const bool mine = (l15 == (bi & 15));
            #pragma unroll
            for (int nt = 0; nt < NT_TILES; ++nt)
                if (mine && nt == nsel) dl[nt] = BIGF;
        }
    }
}

extern "C" void kernel_launch(void* const* d_in, const int* in_sizes, int n_in,
                              void* d_out, int out_size, void* d_ws, size_t ws_size,
                              hipStream_t stream) {
    const float* x   = (const float*)d_in[0];   // [16384, 256] fp32
    const float* ref = (const float*)d_in[1];   // [100, 256] fp32
    int* out = (int*)d_out;                     // [16384, 5] int32 indices
    (void)in_sizes; (void)n_in; (void)out_size; (void)d_ws; (void)ws_size;

    dim3 grid(ROWS / 128);                      // 8 waves * 16 rows per block
    dim3 block(256);
    hipLaunchKernelGGL(knn_topk_kernel, grid, block, SMEM_BYTES, stream,
                       x, ref, out);
}
